// KPConv_14920716386526
// MI455X (gfx1250) — matrix operand order repaired and checked by reading, not compiled, not run on hardware
//
#include <hip/hip_runtime.h>
#include <hip/hip_bf16.h>

typedef __attribute__((ext_vector_type(16))) __bf16 v16bf;
typedef __attribute__((ext_vector_type(8)))  float  v8f;

#define KPTS   27
#define NBRS   32
#define CIN    64
#define COUT   64
#define K2DIM  (KPTS * CIN)    // 1728
#define KSTEPS (K2DIM / 32)    // 54

__device__ __forceinline__ __bf16 f2bf(float f) { return (__bf16)f; }

// ---------------------------------------------------------------------------
// Prep: repack weights [27,64,64] f32 (== [1728,64] row-major) into bf16
// B-fragment order for v_wmma_f32_16x16x32_bf16 (wave32):
//   element e = (((ntile*54 + kstep)*32 + lane)*16 + idx)
//   K = kstep*32 + (lane>>4)*16 + idx ; n = ntile*16 + (lane&15)
//   value = Wflat[K*64 + n]
// ---------------------------------------------------------------------------
__global__ void kpconv_prep_weights(const float* __restrict__ w,
                                    unsigned short* __restrict__ wf) {
    int e = blockIdx.x * blockDim.x + threadIdx.x;
    const int TOT = 4 * KSTEPS * 32 * 16;
    if (e >= TOT) return;
    int idx   = e & 15;
    int lane  = (e >> 4) & 31;
    int kstep = (e >> 9) % KSTEPS;
    int ntile = e / (KSTEPS * 512);
    int K = kstep * 32 + ((lane >> 4) * 16) + idx;
    int n = ntile * 16 + (lane & 15);
    __bf16 v = f2bf(w[(size_t)K * COUT + n]);
    wf[e] = __builtin_bit_cast(unsigned short, v);
}

// ---------------------------------------------------------------------------
// Main: one 128-thread workgroup (4 waves) per tile of 16 query points.
// Phase 1 (per query, whole WG cooperates):
//   gather 32 neighbor features -> featT[64c][32h] (bf16, LDS)
//   compute 27 kernel-point weights per neighbor -> wAT[32l][32h] (bf16, LDS)
//   G[l,c] = Wt x F via 8 wmma tiles (2 mtiles x 4 ntiles) -> GL rows (bf16)
//   (pad rows l>=27 are redirected branch-free to a dump row, GL row 16)
// Phase 2 (per tile):
//   fx[16q,64o] = GL[16 x 1728] x Wfrag[1728 x 64], 54 k-step wmma per wave
// ---------------------------------------------------------------------------
__global__ __launch_bounds__(128)
void kpconv_main(const float* __restrict__ q_pts, const float* __restrict__ s_pts,
                 const int* __restrict__ nbr,     const float* __restrict__ x,
                 const unsigned short* __restrict__ wfrag, float* __restrict__ out,
                 int N, int S) {
    __shared__ __align__(16) __bf16 wAT[32][32];          // A: rows=l (pad 32), cols=h
    __shared__ __align__(16) __bf16 featT[CIN][NBRS];     // B: rows=c, cols=h
    __shared__ __align__(16) __bf16 GLf[17 * K2DIM];      // rows 0..15 = queries, 16 = dump

    const int tid  = threadIdx.x;
    const int wave = tid >> 5;
    const int lane = tid & 31;
    const int tile = blockIdx.x;

    // zero the padded kernel-point rows once (read after first barrier)
    if (tid < 32) {
        for (int l = KPTS; l < 32; ++l) wAT[l][tid] = (__bf16)0.0f;
    }

    // Warm the cache for this wave's step-2 weight slice (54 KB) while we do
    // phase 1: speculative prefetch, 13 x 32 lanes x 128B lines ~= 52 KB.
    {
        const char* base = (const char*)(wfrag + (size_t)wave * KSTEPS * 512);
        #pragma unroll
        for (int p = 0; p < 13; ++p)
            __builtin_prefetch(base + ((size_t)(p * 32 + lane)) * 128, 0, 0);
    }

    const float s_kp    = 0.057735026919f;        // 0.1 / sqrt(3)
    const float inv_ext = 11.547005383792515f;    // 1 / KP_EXTENT

    for (int ql = 0; ql < 16; ++ql) {
        int  n  = tile * 16 + ql;
        bool vq = (n < N);
        int  nq = vq ? n : 0;

        // ---- gather + influence weights ----
        {
            int h     = tid >> 2;        // 4 threads per neighbor
            int chunk = tid & 3;         // 16 channels each
            int ind   = nbr[(size_t)nq * NBRS + h];
            bool vi   = vq && ((unsigned)ind < (unsigned)S);
            int indc  = vi ? ind : 0;
            const float4* xp = (const float4*)(x + (size_t)indc * CIN + chunk * 16);
            #pragma unroll
            for (int v = 0; v < 4; ++v) {
                float4 f = vi ? xp[v] : make_float4(0.f, 0.f, 0.f, 0.f);
                int c = chunk * 16 + v * 4;
                featT[c + 0][h] = f2bf(f.x);
                featT[c + 1][h] = f2bf(f.y);
                featT[c + 2][h] = f2bf(f.z);
                featT[c + 3][h] = f2bf(f.w);
            }
            if (tid < NBRS) {            // one thread per wAT column tid; neighbor h = tid >> 2
                float qx = q_pts[(size_t)nq * 3 + 0];
                float qy = q_pts[(size_t)nq * 3 + 1];
                float qz = q_pts[(size_t)nq * 3 + 2];
                float px, py, pz;
                if (vi) {
                    px = s_pts[(size_t)indc * 3 + 0] - qx;
                    py = s_pts[(size_t)indc * 3 + 1] - qy;
                    pz = s_pts[(size_t)indc * 3 + 2] - qz;
                } else { px = 1e6f; py = 1e6f; pz = 1e6f; }
                #pragma unroll
                for (int l = 0; l < KPTS; ++l) {
                    // kp[l] = ( xyz[(l/3)%3], xyz[l/9], xyz[l%3] ) * 0.1/sqrt(3)
                    float kx = (float)((l / 3) % 3 - 1) * s_kp;
                    float ky = (float)((l / 9)     - 1) * s_kp;
                    float kz = (float)((l % 3)     - 1) * s_kp;
                    float dx = px - kx, dy = py - ky, dz = pz - kz;
                    float w = 1.0f - sqrtf(dx * dx + dy * dy + dz * dz) * inv_ext;
                    wAT[l][tid] = f2bf(fmaxf(w, 0.0f));
                }
            }
        }
        __syncthreads();

        // ---- step 1 WMMA: wave -> (mtile, 2 ntiles) ----
        {
            int mtile = wave >> 1;
            int nt0   = (wave & 1) * 2;
            int m  = lane & 15;
            int kb = (lane >> 4) * 8;
            union { v16bf v; uint4 u[2]; } A;
            A.u[0] = *(const uint4*)&wAT[mtile * 16 + m][kb];
            A.u[1] = *(const uint4*)&wAT[mtile * 16 + m][16 + kb];
            int hi = lane >> 4;
            #pragma unroll
            for (int bi = 0; bi < 2; ++bi) {
                int ntile = nt0 + bi;
                int c   = ntile * 16 + (lane & 15);
                int kb2 = (lane >> 4) * 8;
                union { v16bf v; uint4 u[2]; } B;
                B.u[0] = *(const uint4*)&featT[c][kb2];
                B.u[1] = *(const uint4*)&featT[c][16 + kb2];
                v8f cz = {};
                v8f D = __builtin_amdgcn_wmma_f32_16x16x32_bf16(
                            false, A.v, false, B.v, (short)0, cz, false, false);
                // Branch-free store: out-of-range kernel points (lk >= 27) go
                // to the dump row (row 16) via address select, no exec masking.
                #pragma unroll
                for (int j = 0; j < 8; ++j) {
                    int lk    = mtile * 16 + j + 8 * hi;       // kernel-point index
                    int offV  = ql * K2DIM + lk * CIN + c;     // valid slot
                    int offD  = 16 * K2DIM + c;                // dump slot
                    int off   = (lk < KPTS) ? offV : offD;
                    GLf[off] = f2bf(D[j]);
                }
            }
        }
        __syncthreads();
    }

    // ---- step 2 WMMA: [16 x 1728] x [1728 x 64], wave owns one 16-col ntile ----
    {
        int ntile = wave;
        v8f acc = {};
        const unsigned short* wf = wfrag + (size_t)ntile * KSTEPS * 512;
        int q  = lane & 15;
        int kb = (lane >> 4) * 16;
        #pragma unroll 6
        for (int ks = 0; ks < KSTEPS; ++ks) {
            union { v16bf v; uint4 u[2]; } A;
            A.u[0] = *(const uint4*)&GLf[q * K2DIM + ks * 32 + kb];
            A.u[1] = *(const uint4*)&GLf[q * K2DIM + ks * 32 + kb + 8];
            union { v16bf v; uint4 u[2]; } B;
            const uint4* bp = (const uint4*)(wf + (size_t)(ks * 32 + lane) * 16);
            B.u[0] = bp[0];
            B.u[1] = bp[1];
            acc = __builtin_amdgcn_wmma_f32_16x16x32_bf16(
                      false, A.v, false, B.v, (short)0, acc, false, false);
        }
        int c  = ntile * 16 + (lane & 15);
        int hi = lane >> 4;
        if (tile * 16 + 15 < N) {
            // fast path: whole tile in range, unconditional stores
            #pragma unroll
            for (int j = 0; j < 8; ++j) {
                int n = tile * 16 + j + 8 * hi;
                out[(size_t)n * COUT + c] = acc[j];
            }
        } else {
            #pragma unroll
            for (int j = 0; j < 8; ++j) {
                int n = tile * 16 + j + 8 * hi;
                if (n < N) out[(size_t)n * COUT + c] = acc[j];
            }
        }
    }
}

// ---------------------------------------------------------------------------
extern "C" void kernel_launch(void* const* d_in, const int* in_sizes, int n_in,
                              void* d_out, int out_size, void* d_ws, size_t ws_size,
                              hipStream_t stream) {
    const float* q_pts = (const float*)d_in[0];
    const float* s_pts = (const float*)d_in[1];
    const int*   nbr   = (const int*)d_in[2];
    const float* x     = (const float*)d_in[3];
    const float* w     = (const float*)d_in[4];
    int N = in_sizes[0] / 3;
    int S = in_sizes[1] / 3;

    unsigned short* wfrag = (unsigned short*)d_ws;   // 1728*64*2 = 221184 B
    const int totw = 4 * KSTEPS * 32 * 16;
    kpconv_prep_weights<<<(totw + 255) / 256, 256, 0, stream>>>(w, wfrag);

    int tiles = (N + 15) / 16;
    kpconv_main<<<tiles, 128, 0, stream>>>(q_pts, s_pts, nbr, x, wfrag,
                                           (float*)d_out, N, S);
}
